// VSSBlock_21784074125698
// MI455X (gfx1250) — matrix operand order repaired and checked by reading, not compiled, not run on hardware
//
#include <hip/hip_runtime.h>
#include <hip/hip_bf16.h>

// ---------------------------------------------------------------------------
// CDNA5 (gfx1250) VSS/VMamba block.
// Precision: bf16 WMMA (v_wmma_f32_16x16x32_bf16) for all K=256 GEMMs,
// f32 everywhere else. 3-pass chunked parallel selective scan.
// GEMM v3: A+B staged in padded LDS via b128 copies; all fragments of a
// k-step loaded into distinct registers before a back-to-back wmma chain.
// ---------------------------------------------------------------------------

typedef __attribute__((ext_vector_type(16))) __bf16 v16bf;
typedef __attribute__((ext_vector_type(8)))  float  v8f;
typedef unsigned int u32x4 __attribute__((ext_vector_type(4)));
typedef float        f32x4 __attribute__((ext_vector_type(4)));

#define CB   256     // channels
#define BB   4       // batch
#define LL   4096    // H*W
#define SS   16      // 4*B sequences
#define NCHK 32      // scan chunks
#define CHL  128     // chunk length

__device__ __forceinline__ unsigned short f2bf(float f) {
  unsigned int u = __float_as_uint(f);
  unsigned int r = u + 0x7FFFu + ((u >> 16) & 1u);
  return (unsigned short)(r >> 16);
}
__device__ __forceinline__ float bf2f(unsigned short h) {
  return __uint_as_float(((unsigned int)h) << 16);
}

union FragU { u32x4 q[2]; v16bf v; };

// A fragment: 16x32 bf16; lane holds row (lane&15), halves {k..k+7, k+16..k+23}
__device__ __forceinline__ v16bf ldfragA(const unsigned short* row, int k) {
  FragU f;
  f.q[0] = *(const u32x4*)(row + k);
  f.q[1] = *(const u32x4*)(row + k + 16);
  return f.v;
}
// B fragment: 32x16 bf16; lane holds col (lane&15), 16 contiguous K halves
__device__ __forceinline__ v16bf ldfragB(const unsigned short* p) {
  FragU f;
  f.q[0] = *(const u32x4*)(p);
  f.q[1] = *(const u32x4*)(p + 8);
  return f.v;
}
__device__ __forceinline__ v8f wmma_bf16(v16bf a, v16bf b, v8f c) {
  return __builtin_amdgcn_wmma_f32_16x16x32_bf16(false, a, false, b, (short)0, c,
                                                 false, false);
}

// ---------------------------------------------------------------------------
// Weight prep: fp32 -> bf16 weight copies, A = -exp(Alog)
// ---------------------------------------------------------------------------
__global__ void __launch_bounds__(256) prep_weights_kernel(
    const float* __restrict__ inpj, const float* __restrict__ minw,
    const float* __restrict__ xpj, const float* __restrict__ moutw,
    const float* __restrict__ opj, const float* __restrict__ alog,
    unsigned short* __restrict__ inb, unsigned short* __restrict__ minb,
    unsigned short* __restrict__ xpb, unsigned short* __restrict__ mob,
    unsigned short* __restrict__ opb, float* __restrict__ Aneg) {
  int i = blockIdx.x * 256 + threadIdx.x;
  if (i < 65536)  inb[i] = f2bf(inpj[i]);
  if (i < 131072) minb[i] = f2bf(minw[i]);
  if (i < 16384) { int r = i >> 8; xpb[i] = (r < 24) ? f2bf(xpj[i]) : (unsigned short)0; }
  if (i < 65536)  mob[i] = f2bf(moutw[i]);
  if (i < 65536)  opb[i] = f2bf(opj[i]);
  if (i < 1024)   Aneg[i] = -__expf(alog[i]);
}

// ---------------------------------------------------------------------------
// NCHW -> (B,L,C) transpose (LDS tiled)
// ---------------------------------------------------------------------------
__global__ void __launch_bounds__(256) nchw_to_blc_kernel(
    const float* __restrict__ x, float* __restrict__ xf) {
  __shared__ float t[32][33];
  int b = blockIdx.z;
  int l0 = blockIdx.x * 32, c0 = blockIdx.y * 32;
  int tx = threadIdx.x, ty = threadIdx.y;
#pragma unroll
  for (int j = 0; j < 4; ++j)
    t[ty + j * 8][tx] = x[((size_t)b * CB + c0 + ty + j * 8) * LL + l0 + tx];
  __syncthreads();
#pragma unroll
  for (int j = 0; j < 4; ++j)
    xf[((size_t)b * LL + l0 + ty + j * 8) * CB + c0 + tx] = t[tx][ty + j * 8];
}

// (B,L,C) residual add -> NCHW
__global__ void __launch_bounds__(256) blc_to_nchw_add_kernel(
    const float* __restrict__ xf, const float* __restrict__ hout,
    float* __restrict__ out) {
  __shared__ float t[32][33];
  int b = blockIdx.z;
  int l0 = blockIdx.x * 32, c0 = blockIdx.y * 32;
  int tx = threadIdx.x, ty = threadIdx.y;
#pragma unroll
  for (int j = 0; j < 4; ++j) {
    size_t i = ((size_t)b * LL + l0 + ty + j * 8) * CB + c0 + tx;
    t[ty + j * 8][tx] = xf[i] + hout[i];
  }
  __syncthreads();
#pragma unroll
  for (int j = 0; j < 4; ++j)
    out[((size_t)b * CB + c0 + ty + j * 8) * LL + l0 + tx] = t[tx][ty + j * 8];
}

// ---------------------------------------------------------------------------
// LayerNorm over C=256, one wave per row, bf16 output
// ---------------------------------------------------------------------------
__global__ void __launch_bounds__(256) ln_rows_kernel(
    const float* __restrict__ src, const float* __restrict__ g,
    const float* __restrict__ b, unsigned short* __restrict__ dst) {
  int wv = threadIdx.x >> 5, lane = threadIdx.x & 31;
  size_t row = (size_t)blockIdx.x * 8 + wv;
  const float* p = src + row * CB + lane * 8;
  f32x4 a0 = *(const f32x4*)p;
  f32x4 a1 = *(const f32x4*)(p + 4);
  float v[8] = {a0.x, a0.y, a0.z, a0.w, a1.x, a1.y, a1.z, a1.w};
  float s = 0.f;
#pragma unroll
  for (int i = 0; i < 8; ++i) s += v[i];
#pragma unroll
  for (int m = 16; m >= 1; m >>= 1) s += __shfl_xor(s, m, 32);
  float mu = s * (1.f / 256.f);
  float vs = 0.f;
#pragma unroll
  for (int i = 0; i < 8; ++i) { float d = v[i] - mu; vs += d * d; }
#pragma unroll
  for (int m = 16; m >= 1; m >>= 1) vs += __shfl_xor(vs, m, 32);
  float rstd = rsqrtf(vs * (1.f / 256.f) + 1e-5f);
  int c0 = lane * 8;
#pragma unroll
  for (int i = 0; i < 8; ++i) {
    float o = (v[i] - mu) * rstd * g[c0 + i] + b[c0 + i];
    dst[row * CB + c0 + i] = f2bf(o);
  }
}

// ---------------------------------------------------------------------------
// bf16 WMMA GEMM, K=256: out[M,N] = Xb[M,256] @ Wb[N,256]^T
// Block: 256 thr (8 waves), tile 128 rows x 64 cols.
// X (128x128-half) and W (64x128-half) staged per-kc in padded LDS with b128
// copies; per k-step: load A + all 4 B fragments into distinct registers,
// then 4 back-to-back v_wmma.
// Epilogue: cols < nsplit -> fp32 outF; else -> bf16 silu outS. cols>=storeN skipped.
// ---------------------------------------------------------------------------
__global__ void __launch_bounds__(256) gemm_bf16_k256_kernel(
    const unsigned short* __restrict__ Xb, const unsigned short* __restrict__ Wb,
    float* __restrict__ outF, int ldF, unsigned short* __restrict__ outS, int ldS,
    int nsplit, int storeN) {
  __shared__ unsigned short X[128 * 136];  // 34,816 B; stride 272B (16B-aligned, bank-staggered)
  __shared__ unsigned short Wt[64 * 136];  // 17,408 B; same padded layout
  const int tid = threadIdx.x;
  const size_t m0 = (size_t)blockIdx.x * 128;
  const int n0 = blockIdx.y * 64;
  const int wv = tid >> 5, lane = tid & 31;
  const int r0 = wv * 16;
  const int lrow = lane & 15, hi = lane >> 4, off = hi * 8;
  const unsigned short* arow = X + (r0 + lrow) * 136;
  const unsigned short* w0 = Wt + (0  + lrow) * 136;
  const unsigned short* w1 = Wt + (16 + lrow) * 136;
  const unsigned short* w2 = Wt + (32 + lrow) * 136;
  const unsigned short* w3 = Wt + (48 + lrow) * 136;

  v8f zero = {0.f, 0.f, 0.f, 0.f, 0.f, 0.f, 0.f, 0.f};
  v8f acc[4];
  acc[0] = zero; acc[1] = zero; acc[2] = zero; acc[3] = zero;

  const u32x4* srcX4 = (const u32x4*)(Xb + m0 * CB);        // 32 chunks per row
  const u32x4* srcW4 = (const u32x4*)(Wb + (size_t)n0 * CB);
  unsigned int* Xd = (unsigned int*)X;
  unsigned int* Wd = (unsigned int*)Wt;

#pragma unroll
  for (int kc = 0; kc < 2; ++kc) {
    __syncthreads();
    // Stage X half-slab: 128 rows x 16 b128-chunks = 2048 chunks
#pragma unroll
    for (int i = 0; i < 8; ++i) {
      int flat = i * 256 + tid;
      int r = flat >> 4, c4 = flat & 15;
      *(u32x4*)(Xd + r * 68 + c4 * 4) = srcX4[r * 32 + kc * 16 + c4];
    }
    // Stage W half-slab: 64 rows x 16 b128-chunks = 1024 chunks
#pragma unroll
    for (int i = 0; i < 4; ++i) {
      int flat = i * 256 + tid;
      int r = flat >> 4, c4 = flat & 15;
      *(u32x4*)(Wd + r * 68 + c4 * 4) = srcW4[r * 32 + kc * 16 + c4];
    }
    __syncthreads();
#pragma unroll
    for (int kb = 0; kb < 4; ++kb) {
      int kk = kb * 32 + hi * 16;
      v16bf a  = ldfragB(arow + kk);
      v16bf b0 = ldfragB(w0 + kk);
      v16bf b1 = ldfragB(w1 + kk);
      v16bf b2 = ldfragB(w2 + kk);
      v16bf b3 = ldfragB(w3 + kk);
      acc[0] = wmma_bf16(a, b0, acc[0]);
      acc[1] = wmma_bf16(a, b1, acc[1]);
      acc[2] = wmma_bf16(a, b2, acc[2]);
      acc[3] = wmma_bf16(a, b3, acc[3]);
    }
  }
#pragma unroll
  for (int t = 0; t < 4; ++t) {
    int col = n0 + t * 16 + lrow;
    if (col >= storeN) continue;
#pragma unroll
    for (int v = 0; v < 8; ++v) {
      size_t row = m0 + r0 + v + hi * 8;
      float val = acc[t][v];
      if (col < nsplit) {
        outF[row * (size_t)ldF + col] = val;
      } else {
        float sg = val / (1.f + __expf(-val));
        outS[row * (size_t)ldS + (col - nsplit)] = f2bf(sg);
      }
    }
  }
}

// ---------------------------------------------------------------------------
// Depthwise 3x3 conv (pad 1) on (B,L,C) + SiLU -> bf16
// ---------------------------------------------------------------------------
__global__ void __launch_bounds__(256) dwconv_silu_kernel(
    const float* __restrict__ h, const float* __restrict__ w9,
    unsigned short* __restrict__ out) {
  int c = threadIdx.x;
  int bl = blockIdx.x;
  int b = bl >> 12, l = bl & 4095;
  int y = l >> 6, x = l & 63;
  float acc = 0.f;
#pragma unroll
  for (int dy = -1; dy <= 1; ++dy) {
    int yy = y + dy;
    if ((unsigned)yy >= 64u) continue;
#pragma unroll
    for (int dx = -1; dx <= 1; ++dx) {
      int xx = x + dx;
      if ((unsigned)xx >= 64u) continue;
      acc += h[((size_t)b * LL + yy * 64 + xx) * CB + c] *
             w9[c * 9 + (dy + 1) * 3 + (dx + 1)];
    }
  }
  float s = acc / (1.f + __expf(-acc));
  out[((size_t)b * LL + l) * CB + c] = f2bf(s);
}

// ---------------------------------------------------------------------------
// Build 4 scan directions (bf16 row gather)
// ---------------------------------------------------------------------------
__global__ void __launch_bounds__(256) build_xs4_kernel(
    const unsigned short* __restrict__ hs, unsigned short* __restrict__ xs4) {
  int t = threadIdx.x;
  int l = blockIdx.x;
  int y = blockIdx.y;          // dir*4 + b
  int dir = y >> 2, b = y & 3;
  int ls;
  if (dir == 0)      ls = l;
  else if (dir == 1) ls = LL - 1 - l;
  else if (dir == 2) ls = (l & 63) * 64 + (l >> 6);
  else             { int lt = LL - 1 - l; ls = (lt & 63) * 64 + (lt >> 6); }
  xs4[((size_t)y * LL + l) * CB + t] = hs[((size_t)b * LL + ls) * CB + t];
}

// ---------------------------------------------------------------------------
// Causal depthwise conv1d (k=3) + bias + SiLU -> fp32 + bf16
// ---------------------------------------------------------------------------
__global__ void __launch_bounds__(256) conv1d_silu_kernel(
    const float* __restrict__ xin, const float* __restrict__ w3,
    const float* __restrict__ bias, float* __restrict__ xs,
    unsigned short* __restrict__ xsb) {
  int c = threadIdx.x;
  int sl = blockIdx.x;
  int s = sl >> 12, l = sl & 4095;
  float acc = bias[c];
#pragma unroll
  for (int k = 0; k < 3; ++k) {
    int ll = l - 2 + k;
    if (ll >= 0) acc += xin[((size_t)s * LL + ll) * CB + c] * w3[c * 3 + k];
  }
  float v = acc / (1.f + __expf(-acc));
  size_t i = (size_t)sl * CB + c;
  xs[i] = v;
  xsb[i] = f2bf(v);
}

// ---------------------------------------------------------------------------
// delta = softplus(dt @ dt_w^T + dt_b), dt = dbl[:, :16]
// ---------------------------------------------------------------------------
__global__ void __launch_bounds__(256) delta_kernel(
    const float* __restrict__ dbl, const float* __restrict__ dtw,
    const float* __restrict__ dtb, float* __restrict__ delta) {
  __shared__ float dt[16];
  int c = threadIdx.x;
  size_t m = blockIdx.x;
  if (c < 16) dt[c] = dbl[m * 24 + c];
  __syncthreads();
  float acc = dtb[c];
  const float* wr = dtw + c * 16;
#pragma unroll
  for (int r = 0; r < 16; ++r) acc += dt[r] * wr[r];
  float d = (acc > 20.f) ? acc : __logf(1.f + __expf(acc));
  delta[m * CB + c] = d;
}

// ---------------------------------------------------------------------------
// Selective scan, pass A: per-chunk local scan; emit y_local, (prod dA, h_T)
// ---------------------------------------------------------------------------
__global__ void __launch_bounds__(256) scan_partial_kernel(
    const float* __restrict__ delta, const float* __restrict__ xs,
    const float* __restrict__ dbl, const float* __restrict__ Aneg,
    float* __restrict__ y, float* __restrict__ PH) {
  __shared__ float bc[CHL][8];  // cols 16..23 of dbl: B (4) then C (4)
  int d = threadIdx.x;
  int ch = blockIdx.x, s = blockIdx.y;
  int l0 = ch * CHL;
#pragma unroll
  for (int i = 0; i < 4; ++i) {
    int flat = i * 256 + d;
    int l = flat >> 3, j = flat & 7;
    bc[l][j] = dbl[((size_t)s * LL + l0 + l) * 24 + 16 + j];
  }
  __syncthreads();
  float A0 = Aneg[d * 4 + 0], A1 = Aneg[d * 4 + 1];
  float A2 = Aneg[d * 4 + 2], A3 = Aneg[d * 4 + 3];
  float h0 = 0, h1 = 0, h2 = 0, h3 = 0;
  float p0 = 1, p1 = 1, p2 = 1, p3 = 1;
  for (int l = 0; l < CHL; ++l) {
    size_t idx = ((size_t)s * LL + l0 + l) * CB + d;
    float dlt = delta[idx];
    float u = xs[idx];
    __builtin_prefetch(delta + idx + CB, 0, 1);
    __builtin_prefetch(xs + idx + CB, 0, 1);
    float du = dlt * u;
    float e0 = __expf(dlt * A0), e1 = __expf(dlt * A1);
    float e2 = __expf(dlt * A2), e3 = __expf(dlt * A3);
    h0 = e0 * h0 + du * bc[l][0];
    h1 = e1 * h1 + du * bc[l][1];
    h2 = e2 * h2 + du * bc[l][2];
    h3 = e3 * h3 + du * bc[l][3];
    p0 *= e0; p1 *= e1; p2 *= e2; p3 *= e3;
    y[idx] = h0 * bc[l][4] + h1 * bc[l][5] + h2 * bc[l][6] + h3 * bc[l][7];
  }
  size_t base = (((size_t)s * NCHK + ch) * 8) * CB + d;
  PH[base + 0 * CB] = p0; PH[base + 1 * CB] = p1;
  PH[base + 2 * CB] = p2; PH[base + 3 * CB] = p3;
  PH[base + 4 * CB] = h0; PH[base + 5 * CB] = h1;
  PH[base + 6 * CB] = h2; PH[base + 7 * CB] = h3;
}

// Pass B: serial prefix over chunks -> incoming state per chunk
__global__ void __launch_bounds__(256) scan_combine_kernel(
    const float* __restrict__ PH, float* __restrict__ HIN) {
  int d = threadIdx.x, s = blockIdx.x;
  float h0 = 0, h1 = 0, h2 = 0, h3 = 0;
  for (int ch = 0; ch < NCHK; ++ch) {
    size_t hb = (((size_t)s * NCHK + ch) * 4) * CB + d;
    HIN[hb + 0 * CB] = h0; HIN[hb + 1 * CB] = h1;
    HIN[hb + 2 * CB] = h2; HIN[hb + 3 * CB] = h3;
    size_t pb = (((size_t)s * NCHK + ch) * 8) * CB + d;
    h0 = PH[pb + 0 * CB] * h0 + PH[pb + 4 * CB];
    h1 = PH[pb + 1 * CB] * h1 + PH[pb + 5 * CB];
    h2 = PH[pb + 2 * CB] * h2 + PH[pb + 6 * CB];
    h3 = PH[pb + 3 * CB] * h3 + PH[pb + 7 * CB];
  }
}

// Pass C: propagate incoming state through chunk, fix up y
__global__ void __launch_bounds__(256) scan_fixup_kernel(
    const float* __restrict__ delta, const float* __restrict__ dbl,
    const float* __restrict__ Aneg, const float* __restrict__ HIN,
    float* __restrict__ y) {
  int d = threadIdx.x;
  int ch = blockIdx.x, s = blockIdx.y;
  if (ch == 0) return;  // incoming state is zero
  __shared__ float cs[CHL][4];
  int l0 = ch * CHL;
#pragma unroll
  for (int i = 0; i < 2; ++i) {
    int flat = i * 256 + d;
    int l = flat >> 2, j = flat & 3;
    cs[l][j] = dbl[((size_t)s * LL + l0 + l) * 24 + 20 + j];
  }
  __syncthreads();
  size_t hb = (((size_t)s * NCHK + ch) * 4) * CB + d;
  float h0 = HIN[hb + 0 * CB], h1 = HIN[hb + 1 * CB];
  float h2 = HIN[hb + 2 * CB], h3 = HIN[hb + 3 * CB];
  float A0 = Aneg[d * 4 + 0], A1 = Aneg[d * 4 + 1];
  float A2 = Aneg[d * 4 + 2], A3 = Aneg[d * 4 + 3];
  for (int l = 0; l < CHL; ++l) {
    size_t idx = ((size_t)s * LL + l0 + l) * CB + d;
    float dlt = delta[idx];
    __builtin_prefetch(delta + idx + CB, 0, 1);
    h0 *= __expf(dlt * A0);
    h1 *= __expf(dlt * A1);
    h2 *= __expf(dlt * A2);
    h3 *= __expf(dlt * A3);
    y[idx] += h0 * cs[l][0] + h1 * cs[l][1] + h2 * cs[l][2] + h3 * cs[l][3];
  }
}

// ---------------------------------------------------------------------------
// ymod = (y + xs*Dp) * silu(z)   (silu(z) already stored as bf16)
// ---------------------------------------------------------------------------
__global__ void __launch_bounds__(256) ymod_kernel(
    const float* __restrict__ y, const float* __restrict__ xs,
    const float* __restrict__ Dp, const unsigned short* __restrict__ zs,
    unsigned short* __restrict__ out) {
  int c = threadIdx.x;
  size_t m = blockIdx.x;
  size_t i = m * CB + c;
  float v = (y[i] + xs[i] * Dp[c]) * bf2f(zs[i]);
  out[i] = f2bf(v);
}

// ---------------------------------------------------------------------------
// Combine 4 directions -> yf (avg/4)
// ---------------------------------------------------------------------------
__global__ void __launch_bounds__(256) combine_kernel(
    const float* __restrict__ y4, float* __restrict__ yf) {
  int c = threadIdx.x;
  int l = blockIdx.x, b = blockIdx.y;
  int lt = (l & 63) * 64 + (l >> 6);
  float v = y4[(((size_t)0  + b) * LL + l) * CB + c]
          + y4[(((size_t)4  + b) * LL + (LL - 1 - l)) * CB + c]
          + y4[(((size_t)8  + b) * LL + lt) * CB + c]
          + y4[(((size_t)12 + b) * LL + (LL - 1 - lt)) * CB + c];
  yf[((size_t)b * LL + l) * CB + c] = 0.25f * v;
}

// ---------------------------------------------------------------------------
// Host launcher
// ---------------------------------------------------------------------------
extern "C" void kernel_launch(void* const* d_in, const int* in_sizes, int n_in,
                              void* d_out, int out_size, void* d_ws, size_t ws_size,
                              hipStream_t stream) {
  const float* x          = (const float*)d_in[0];
  const float* ln1_g      = (const float*)d_in[1];
  const float* ln1_b      = (const float*)d_in[2];
  const float* in_proj_w  = (const float*)d_in[3];
  const float* dw_w       = (const float*)d_in[4];
  const float* m_in_w     = (const float*)d_in[5];
  const float* m_conv_w   = (const float*)d_in[6];
  const float* m_conv_b   = (const float*)d_in[7];
  const float* m_xproj_w  = (const float*)d_in[8];
  const float* m_dt_w     = (const float*)d_in[9];
  const float* m_dt_b     = (const float*)d_in[10];
  const float* m_Alog     = (const float*)d_in[11];
  const float* m_Dp       = (const float*)d_in[12];
  const float* m_out_w    = (const float*)d_in[13];
  const float* lnss_g     = (const float*)d_in[14];
  const float* lnss_b     = (const float*)d_in[15];
  const float* out_proj_w = (const float*)d_in[16];
  float* out = (float*)d_out;

  const size_t BLC  = (size_t)BB * LL * CB;   // 4,194,304
  const size_t B4LC = 4 * BLC;                // 16,777,216

  char* w = (char*)d_ws;
  size_t off = 0;
  auto take = [&](size_t bytes) -> char* {
    char* p = w + off;
    off += (bytes + 255) & ~(size_t)255;
    return p;
  };
  unsigned short* inprojb = (unsigned short*)take(65536 * 2);
  unsigned short* minb    = (unsigned short*)take(131072 * 2);
  unsigned short* xprojb  = (unsigned short*)take(16384 * 2);   // 64x256, rows>=24 zero
  unsigned short* moutb   = (unsigned short*)take(65536 * 2);
  unsigned short* oprojb  = (unsigned short*)take(65536 * 2);
  float* Aneg   = (float*)take(1024 * 4);
  float* xf     = (float*)take(BLC * 4);
  unsigned short* x1n = (unsigned short*)take(BLC * 2);
  float* hbuf   = (float*)take(BLC * 4);             // in_proj out; reused as hout
  unsigned short* hs  = (unsigned short*)take(BLC * 2);
  unsigned short* xs4 = (unsigned short*)take(B4LC * 2);  // reused as ymod
  float* xs_pre = (float*)take(B4LC * 4);            // reused as yscan
  unsigned short* zsil = (unsigned short*)take(B4LC * 2);
  float* xs     = (float*)take(B4LC * 4);            // reused as y4
  unsigned short* xsb = (unsigned short*)take(B4LC * 2);
  float* dbl    = (float*)take((size_t)SS * LL * 24 * 4);
  float* delta  = (float*)take(B4LC * 4);
  float* PH     = (float*)take((size_t)SS * NCHK * 8 * CB * 4);
  float* HIN    = (float*)take((size_t)SS * NCHK * 4 * CB * 4);
  float* yf     = (float*)take(BLC * 4);
  unsigned short* yfn = (unsigned short*)take(BLC * 2);
  float* yscan = xs_pre;        // alias (xs_pre dead after conv1d)
  unsigned short* ymod = xs4;   // alias (xs4 dead after GEMM2)
  float* y4 = xs;               // alias (xs dead after ymod)
  float* hout = hbuf;           // alias (h dead after dwconv)

  // 0. weights -> bf16, A = -exp(Alog)
  prep_weights_kernel<<<512, 256, 0, stream>>>(in_proj_w, m_in_w, m_xproj_w,
      m_out_w, out_proj_w, m_Alog, inprojb, minb, xprojb, moutb, oprojb, Aneg);
  // 1. NCHW -> (B,L,C)
  nchw_to_blc_kernel<<<dim3(LL / 32, CB / 32, BB), dim3(32, 8), 0, stream>>>(x, xf);
  // 2. LN1 -> bf16
  ln_rows_kernel<<<(BB * LL) / 8, 256, 0, stream>>>(xf, ln1_g, ln1_b, x1n);
  // 3. h = x1n @ in_proj^T
  gemm_bf16_k256_kernel<<<dim3((BB * LL) / 128, 4), 256, 0, stream>>>(
      x1n, inprojb, hbuf, CB, (unsigned short*)nullptr, 0, CB, CB);
  // 4. depthwise 3x3 + silu -> bf16
  dwconv_silu_kernel<<<BB * LL, 256, 0, stream>>>(hbuf, dw_w, hs);
  // 5. 4 scan directions
  build_xs4_kernel<<<dim3(LL, SS), 256, 0, stream>>>(hs, xs4);
  // 6. xz = xs4 @ m_in^T ; split: xs_pre fp32 | silu(z) bf16
  gemm_bf16_k256_kernel<<<dim3((SS * LL) / 128, 8), 256, 0, stream>>>(
      xs4, minb, xs_pre, CB, zsil, CB, CB, 2 * CB);
  // 7. causal conv1d + silu -> xs fp32 + xsb bf16
  conv1d_silu_kernel<<<SS * LL, 256, 0, stream>>>(xs_pre, m_conv_w, m_conv_b, xs, xsb);
  // 8. dbl = xsb @ xproj^T  (N padded to 64, store 24)
  gemm_bf16_k256_kernel<<<dim3((SS * LL) / 128, 1), 256, 0, stream>>>(
      xsb, xprojb, dbl, 24, (unsigned short*)nullptr, 0, 64, 24);
  // 9. delta = softplus(dt @ dt_w^T + dt_b)
  delta_kernel<<<SS * LL, 256, 0, stream>>>(dbl, m_dt_w, m_dt_b, delta);
  // 10-12. chunked parallel selective scan
  scan_partial_kernel<<<dim3(NCHK, SS), 256, 0, stream>>>(delta, xs, dbl, Aneg,
                                                          yscan, PH);
  scan_combine_kernel<<<SS, 256, 0, stream>>>(PH, HIN);
  scan_fixup_kernel<<<dim3(NCHK, SS), 256, 0, stream>>>(delta, dbl, Aneg, HIN,
                                                        yscan);
  // 13. ymod = (y + xs*Dp) * silu(z)  -> bf16
  ymod_kernel<<<SS * LL, 256, 0, stream>>>(yscan, xs, m_Dp, zsil, ymod);
  // 14. y4 = ymod @ m_out^T
  gemm_bf16_k256_kernel<<<dim3((SS * LL) / 128, 4), 256, 0, stream>>>(
      ymod, moutb, y4, CB, (unsigned short*)nullptr, 0, CB, CB);
  // 15. combine 4 directions
  combine_kernel<<<dim3(LL, BB), 256, 0, stream>>>(y4, yf);
  // 16. LNss -> bf16
  ln_rows_kernel<<<(BB * LL) / 8, 256, 0, stream>>>(yf, lnss_g, lnss_b, yfn);
  // 17. hout = yfn @ out_proj^T
  gemm_bf16_k256_kernel<<<dim3((BB * LL) / 128, 4), 256, 0, stream>>>(
      yfn, oprojb, hout, CB, (unsigned short*)nullptr, 0, CB, CB);
  // 18. residual + (B,L,C) -> NCHW
  blc_to_nchw_add_kernel<<<dim3(LL / 32, CB / 32, BB), dim3(32, 8), 0, stream>>>(
      xf, hout, out);
  (void)in_sizes; (void)n_in; (void)out_size; (void)ws_size;
}
